// RotarySelfAttentionBlock_46763603919298
// MI455X (gfx1250) — matrix operand order repaired and checked by reading, not compiled, not run on hardware
//
#include <hip/hip_runtime.h>
#include <hip/hip_bf16.h>

// MI455X / gfx1250 attention block.
// Precision plan: bf16 operands + f32 WMMA accumulation (v_wmma_f32_16x16x32_bf16).
// fp32 WMMA (16x16x4) would leave this compute-bound (~26 G MACs); bf16 WMMA is 8x
// denser and pushes compute below the ~6us HBM/L2 floor (all tensors fit in 192MB L2).
// GEMM tiles are staged with gfx1250 GLOBAL_LOAD_ASYNC_TO_LDS_B128 (ASYNCcnt) and
// double-buffered so the vector pipe only does fragment ds_loads + WMMA.

typedef __attribute__((ext_vector_type(16))) __bf16 bf16x16;
typedef __attribute__((ext_vector_type(8)))  float  floatx8;

union Frag { unsigned int u[8]; bf16x16 v; };
static_assert(sizeof(Frag) == 32, "frag size");

__device__ __forceinline__ unsigned short f2bf(float f) {
  unsigned int u = __float_as_uint(f);
  u += 0x7fffu + ((u >> 16) & 1u);   // round-to-nearest-even
  return (unsigned short)(u >> 16);
}

__device__ __forceinline__ floatx8 wmma_bf16(const Frag& a, const Frag& b, floatx8 c) {
  return __builtin_amdgcn_wmma_f32_16x16x32_bf16(false, a.v, false, b.v,
                                                 (short)0, c, false, false);
}

// 16B-per-lane async copy global -> LDS (GVS mode: SGPR base + 32-bit lane offset).
__device__ __forceinline__ void async_tile16(unsigned lds_addr, unsigned voff,
                                             const unsigned short* sbase) {
  asm volatile("global_load_async_to_lds_b128 %0, %1, %2"
               :: "v"(lds_addr), "v"(voff), "s"(sbase)
               : "memory");
}
__device__ __forceinline__ void wait_async0() {
  asm volatile("s_wait_asynccnt 0x0" ::: "memory");
}

// ---------------------------------------------------------------- converts
__global__ void cvt_f32_bf16(const float* __restrict__ src,
                             unsigned short* __restrict__ dst, int n) {
  int i = blockIdx.x * blockDim.x + threadIdx.x;
  int stride = gridDim.x * blockDim.x;
  for (; i < n; i += stride) dst[i] = f2bf(src[i]);
}

// wt[n][k] = bf16(w[k][n]) : weights become K-contiguous for WMMA B-fragments.
__global__ void transpose_cvt(const float* __restrict__ w,
                              unsigned short* __restrict__ wt, int K, int N) {
  __shared__ float tile[32][33];
  int k0 = blockIdx.x * 32, n0 = blockIdx.y * 32;
  int tx = threadIdx.x, ty = threadIdx.y;   // 32 x 8
#pragma unroll
  for (int i = 0; i < 32; i += 8)
    tile[ty + i][tx] = w[(size_t)(k0 + ty + i) * N + n0 + tx];
  __syncthreads();
#pragma unroll
  for (int i = 0; i < 32; i += 8)
    wt[(size_t)(n0 + ty + i) * K + k0 + tx] = f2bf(tile[tx][ty + i]);
}

// ---------------------------------------------------------------- QKV GEMM + RoPE
// C[4096,3072] = A[4096,1024] * Wt[3072,1024]^T + bias ; epilogue: RoPE on q,k,
// scatter q,k -> [BH,N,D], v -> [BH,D,N] (transposed for P@V B-fragments).
__global__ __launch_bounds__(256) void qkv_gemm_rope(
    const unsigned short* __restrict__ A,
    const unsigned short* __restrict__ Wt,
    const float* __restrict__ bias,
    unsigned short* __restrict__ qb,
    unsigned short* __restrict__ kb,
    unsigned short* __restrict__ vTb) {
  __shared__ unsigned short As[2][64][34];    // pad 34: frag rows hit distinct banks
  __shared__ unsigned short Bs[2][128][34];
  const int tid = threadIdx.x;
  const int lane = tid & 31, wv = tid >> 5;
  const int n = lane & 15, half = lane >> 4;
  const int wrow = (wv & 3) * 16, wcol = (wv >> 2) * 64;
  const int r0 = blockIdx.y * 64;
  const int c0 = blockIdx.x * 128;

  const unsigned short* Abase = A  + (size_t)r0 * 1024;
  const unsigned short* Bbase = Wt + (size_t)c0 * 1024;
  const int lrow = tid >> 2, lc = (tid & 3) * 8;   // 16B chunk owned by this thread

  floatx8 acc[4];
#pragma unroll
  for (int ct = 0; ct < 4; ++ct)
    acc[ct] = floatx8{0.f,0.f,0.f,0.f,0.f,0.f,0.f,0.f};

  auto issue = [&](int kk, int b) {
    async_tile16((unsigned)(uintptr_t)&As[b][lrow][lc],
                 (unsigned)(lrow * 2048 + (kk + lc) * 2), Abase);
#pragma unroll
    for (int it = 0; it < 2; ++it) {
      int idx = tid + it * 256;
      int row = idx >> 2, c = (idx & 3) * 8;
      async_tile16((unsigned)(uintptr_t)&Bs[b][row][c],
                   (unsigned)(row * 2048 + (kk + c) * 2), Bbase);
    }
  };

  issue(0, 0);
  wait_async0();
  __syncthreads();

  for (int i = 0; i < 32; ++i) {
    const int b = i & 1;
    if (i < 31) issue((i + 1) * 32, b ^ 1);   // prefetch next tile into other buffer
    Frag af;
#pragma unroll
    for (int v = 0; v < 8; ++v) {
      int kx = half * 16 + 2 * v;
      af.u[v] = *(const unsigned int*)&As[b][wrow + n][kx];
    }
#pragma unroll
    for (int ct = 0; ct < 4; ++ct) {
      Frag bf;
#pragma unroll
      for (int v = 0; v < 8; ++v) {
        int kx = half * 16 + 2 * v;
        bf.u[v] = *(const unsigned int*)&Bs[b][wcol + ct * 16 + n][kx];
      }
      acc[ct] = wmma_bf16(af, bf, acc[ct]);
    }
    wait_async0();     // own async writes to other buffer have landed in LDS
    __syncthreads();   // everyone done reading buf b / writing buf b^1
  }

  // epilogue: bias + RoPE + scatter. Column block is s-uniform (1024 % 128 == 0).
#pragma unroll
  for (int ct = 0; ct < 4; ++ct) {
    int gc = c0 + wcol + ct * 16 + n;
    int s = gc >> 10, rem = gc & 1023, h = rem >> 6, d = rem & 63;
    float bval = bias[gc];
    // inv_freq = theta^(-(2i)/D), 2i = d&~1, ln(10000)=9.21034037...
    float invf = __expf(-(float)(d & ~1) * (9.210340371976184f / 64.0f));
#pragma unroll
    for (int v = 0; v < 8; ++v) {
      int gr = r0 + wrow + v + 8 * half;
      int bidx = gr >> 10, npos = gr & 1023;
      float val = acc[ct][v] + bval;
      float ang = (float)npos * invf;
      ang -= 6.283185307179586f * floorf(ang * 0.15915494309189535f);
      float sn, cs;
      __sincosf(ang, &sn, &cs);
      float partner = __shfl_xor(val, 1, 32);   // pair (2i, 2i+1) lives in lane n^1
      float r = (d & 1) ? (val * cs + partner * sn) : (val * cs - partner * sn);
      size_t bh = (size_t)(bidx * 16 + h);
      if (s == 0)      qb[(bh * 1024 + npos) * 64 + d] = f2bf(r);
      else if (s == 1) kb[(bh * 1024 + npos) * 64 + d] = f2bf(r);
      else             vTb[(bh * 64 + d) * 1024 + npos] = f2bf(val);
    }
  }
}

// ---------------------------------------------------------------- flash attention
// One wave owns 16 q rows x D=64. Online softmax; P tile bounced through padded LDS
// to convert C-fragment layout -> A-fragment layout for the P@V WMMAs.
__global__ __launch_bounds__(128) void flash_attn(
    const unsigned short* __restrict__ q,
    const unsigned short* __restrict__ k,
    const unsigned short* __restrict__ vT,
    unsigned short* __restrict__ aout) {
  __shared__ unsigned short Pt[4][16][40];   // per-wave 16x32 P tile, padded rows
  const int tid = threadIdx.x;
  const int lane = tid & 31, wv = tid >> 5;
  const int n = lane & 15, half = lane >> 4;
  const int bh = blockIdx.y;
  const int qr0 = blockIdx.x * 64 + wv * 16;
  const unsigned short* qbh = q  + (size_t)bh * 1024 * 64;
  const unsigned short* kbh = k  + (size_t)bh * 1024 * 64;
  const unsigned short* vbh = vT + (size_t)bh * 64 * 1024;

  Frag aq0, aq1;   // q fragments, K = D split 0..31 / 32..63
#pragma unroll
  for (int v = 0; v < 8; ++v) {
    int kx = half * 16 + 2 * v;
    aq0.u[v] = *(const unsigned int*)&qbh[(size_t)(qr0 + n) * 64 + kx];
    aq1.u[v] = *(const unsigned int*)&qbh[(size_t)(qr0 + n) * 64 + 32 + kx];
  }

  floatx8 o[4];
#pragma unroll
  for (int dt = 0; dt < 4; ++dt)
    o[dt] = floatx8{0.f,0.f,0.f,0.f,0.f,0.f,0.f,0.f};
  float rm[8], rl[8];
#pragma unroll
  for (int v = 0; v < 8; ++v) { rm[v] = -3.0e38f; rl[v] = 0.f; }

  for (int kt = 0; kt < 32; ++kt) {          // 32 keys per step
    floatx8 s[2];
#pragma unroll
    for (int t = 0; t < 2; ++t) {
      int ks0 = kt * 32 + t * 16;
      Frag bk0, bk1;
#pragma unroll
      for (int v = 0; v < 8; ++v) {
        int kx = half * 16 + 2 * v;
        bk0.u[v] = *(const unsigned int*)&kbh[(size_t)(ks0 + n) * 64 + kx];
        bk1.u[v] = *(const unsigned int*)&kbh[(size_t)(ks0 + n) * 64 + 32 + kx];
      }
      floatx8 c = floatx8{0.f,0.f,0.f,0.f,0.f,0.f,0.f,0.f};
      c = wmma_bf16(aq0, bk0, c);
      c = wmma_bf16(aq1, bk1, c);
#pragma unroll
      for (int v = 0; v < 8; ++v) s[t][v] = c[v] * 0.125f;   // D^-0.5
    }
    // online softmax: reduce across the 16 lanes of each half-group (wave32)
#pragma unroll
    for (int v = 0; v < 8; ++v) {
      float mt = fmaxf(s[0][v], s[1][v]);
#pragma unroll
      for (int off = 8; off > 0; off >>= 1) mt = fmaxf(mt, __shfl_xor(mt, off, 32));
      float mn = fmaxf(rm[v], mt);
      float alpha = __expf(rm[v] - mn);
      rm[v] = mn;
      float p0 = __expf(s[0][v] - mn);
      float p1 = __expf(s[1][v] - mn);
      s[0][v] = p0; s[1][v] = p1;
      float ps = p0 + p1;
#pragma unroll
      for (int off = 8; off > 0; off >>= 1) ps += __shfl_xor(ps, off, 32);
      rl[v] = rl[v] * alpha + ps;
      o[0][v] *= alpha; o[1][v] *= alpha; o[2][v] *= alpha; o[3][v] *= alpha;
    }
    // C-layout -> LDS (row = v + 8*half, col = lane%16)
#pragma unroll
    for (int v = 0; v < 8; ++v) {
      Pt[wv][v + 8 * half][n]      = f2bf(s[0][v]);
      Pt[wv][v + 8 * half][16 + n] = f2bf(s[1][v]);
    }
    __syncthreads();
    Frag ap;   // re-read as A-fragment over K=32 keys
#pragma unroll
    for (int v = 0; v < 8; ++v) {
      int kx = half * 16 + 2 * v;
      ap.u[v] = *(const unsigned int*)&Pt[wv][n][kx];
    }
#pragma unroll
    for (int dt = 0; dt < 4; ++dt) {
      Frag bv;   // V^T [D][N]: consecutive keys are pair-contiguous
#pragma unroll
      for (int v = 0; v < 8; ++v) {
        bv.u[v] = *(const unsigned int*)
            &vbh[(size_t)(dt * 16 + n) * 1024 + kt * 32 + half * 16 + 2 * v];
      }
      o[dt] = wmma_bf16(ap, bv, o[dt]);
    }
    __syncthreads();
  }

  int b = bh >> 4, h = bh & 15;
#pragma unroll
  for (int v = 0; v < 8; ++v) {
    float inv = 1.0f / rl[v];
    int row = qr0 + v + 8 * half;
#pragma unroll
    for (int dt = 0; dt < 4; ++dt) {
      aout[((size_t)(b * 1024 + row)) * 1024 + h * 64 + dt * 16 + n] =
          f2bf(o[dt][v] * inv);
    }
  }
}

// ---------------------------------------------------------------- output projection
__global__ __launch_bounds__(256) void proj_gemm(
    const unsigned short* __restrict__ A,    // [4096][1024] bf16
    const unsigned short* __restrict__ Wt,   // [1024][1024] bf16 (transposed)
    const float* __restrict__ bias,
    float* __restrict__ out) {               // [4096][1024] f32
  __shared__ unsigned short As[2][64][34];
  __shared__ unsigned short Bs[2][128][34];
  const int tid = threadIdx.x;
  const int lane = tid & 31, wv = tid >> 5;
  const int n = lane & 15, half = lane >> 4;
  const int wrow = (wv & 3) * 16, wcol = (wv >> 2) * 64;
  const int r0 = blockIdx.y * 64;
  const int c0 = blockIdx.x * 128;

  const unsigned short* Abase = A  + (size_t)r0 * 1024;
  const unsigned short* Bbase = Wt + (size_t)c0 * 1024;
  const int lrow = tid >> 2, lc = (tid & 3) * 8;

  floatx8 acc[4];
#pragma unroll
  for (int ct = 0; ct < 4; ++ct)
    acc[ct] = floatx8{0.f,0.f,0.f,0.f,0.f,0.f,0.f,0.f};

  auto issue = [&](int kk, int b) {
    async_tile16((unsigned)(uintptr_t)&As[b][lrow][lc],
                 (unsigned)(lrow * 2048 + (kk + lc) * 2), Abase);
#pragma unroll
    for (int it = 0; it < 2; ++it) {
      int idx = tid + it * 256;
      int row = idx >> 2, c = (idx & 3) * 8;
      async_tile16((unsigned)(uintptr_t)&Bs[b][row][c],
                   (unsigned)(row * 2048 + (kk + c) * 2), Bbase);
    }
  };

  issue(0, 0);
  wait_async0();
  __syncthreads();

  for (int i = 0; i < 32; ++i) {
    const int b = i & 1;
    if (i < 31) issue((i + 1) * 32, b ^ 1);
    Frag af;
#pragma unroll
    for (int v = 0; v < 8; ++v) {
      int kx = half * 16 + 2 * v;
      af.u[v] = *(const unsigned int*)&As[b][wrow + n][kx];
    }
#pragma unroll
    for (int ct = 0; ct < 4; ++ct) {
      Frag bf;
#pragma unroll
      for (int v = 0; v < 8; ++v) {
        int kx = half * 16 + 2 * v;
        bf.u[v] = *(const unsigned int*)&Bs[b][wcol + ct * 16 + n][kx];
      }
      acc[ct] = wmma_bf16(af, bf, acc[ct]);
    }
    wait_async0();
    __syncthreads();
  }

#pragma unroll
  for (int ct = 0; ct < 4; ++ct) {
    int gc = c0 + wcol + ct * 16 + n;
    float bval = bias[gc];
#pragma unroll
    for (int v = 0; v < 8; ++v) {
      int gr = r0 + wrow + v + 8 * half;
      out[(size_t)gr * 1024 + gc] = acc[ct][v] + bval;
    }
  }
}

// ---------------------------------------------------------------- launcher
extern "C" void kernel_launch(void* const* d_in, const int* in_sizes, int n_in,
                              void* d_out, int out_size, void* d_ws, size_t ws_size,
                              hipStream_t stream) {
  (void)in_sizes; (void)n_in; (void)out_size; (void)ws_size;
  const float* x      = (const float*)d_in[0];
  const float* w_qkv  = (const float*)d_in[1];
  const float* b_qkv  = (const float*)d_in[2];
  const float* w_proj = (const float*)d_in[3];
  const float* b_proj = (const float*)d_in[4];
  float* out = (float*)d_out;

  // workspace partition (~51 MB required)
  char* ws = (char*)d_ws;
  size_t off = 0;
  auto alloc = [&](size_t bytes) {
    void* p = ws + off;
    off += (bytes + 255) & ~(size_t)255;
    return p;
  };
  unsigned short* xb     = (unsigned short*)alloc(4096ull * 1024 * 2);
  unsigned short* wqkvT  = (unsigned short*)alloc(3072ull * 1024 * 2);
  unsigned short* wprojT = (unsigned short*)alloc(1024ull * 1024 * 2);
  unsigned short* qb     = (unsigned short*)alloc(64ull * 1024 * 64 * 2);
  unsigned short* kb     = (unsigned short*)alloc(64ull * 1024 * 64 * 2);
  unsigned short* vTb    = (unsigned short*)alloc(64ull * 64 * 1024 * 2);
  unsigned short* aout   = (unsigned short*)alloc(4096ull * 1024 * 2);

  cvt_f32_bf16<<<4096, 256, 0, stream>>>(x, xb, 4096 * 1024);
  transpose_cvt<<<dim3(32, 96), dim3(32, 8), 0, stream>>>(w_qkv, wqkvT, 1024, 3072);
  transpose_cvt<<<dim3(32, 32), dim3(32, 8), 0, stream>>>(w_proj, wprojT, 1024, 1024);
  qkv_gemm_rope<<<dim3(24, 64), 256, 0, stream>>>(xb, wqkvT, b_qkv, qb, kb, vTb);
  flash_attn<<<dim3(16, 64), 128, 0, stream>>>(qb, kb, vTb, aout);
  proj_gemm<<<dim3(8, 64), 256, 0, stream>>>(aout, wprojT, b_proj, out);
}
